// MiniKPCNN_28767690949428
// MI455X (gfx1250) — matrix operand order repaired and checked by reading, not compiled, not run on hardware
//
#include <hip/hip_runtime.h>
#include <hip/hip_bf16.h>

// MiniKPCNN on gfx1250: implicit-GEMM convolutions via v_wmma_f32_16x16x32_f16,
// f16 activations/weights with f32 accumulation, LDS-staged reflect-padded
// input tiles, VALU tail for softmax + kernel-predicting filter bank.
//
// K-axis permutation: input channels are split into two streams (low/high half
// of CIN). Hardware K chunk kc, lane-half h, element t <-> logical tap
// h*Khalf + kc/2 + t. This makes every B-fragment LDS address a compile-time
// immediate offset from one per-lane base (no per-element VALU), while the
// A-side (packed weights) stays contiguous b128 loads. Permutation is applied
// once at weight-pack time.

typedef _Float16 f16;
typedef __attribute__((ext_vector_type(16))) _Float16 v16h;
typedef __attribute__((ext_vector_type(8)))  _Float16 v8h;
typedef __attribute__((ext_vector_type(8)))  float    v8f;

__device__ __forceinline__ int reflect_idx(int i, int n) {
    // jnp.pad mode='reflect' (no edge repeat); pad < n so one fold suffices
    if (i < 0) i = -i;
    if (i >= n) i = 2 * n - 2 - i;
    return i;
}

// ---------------------------------------------------------------------------
// Weight repack: f32 OIHW -> f16 [Coutp][KPAD] in *hardware K order*.
// hw-k = chunk*32 + half*16 + t  ->  logical tap = half*Khalf + chunk*16 + t
// (stream `half` = channel block of CIN/2; OIHW flattening keeps streams
// contiguous). Zero pad for t beyond Khalf and rows beyond Cout.
// ---------------------------------------------------------------------------
__global__ void pack_weights(const float* __restrict__ w, f16* __restrict__ wpk,
                             int Cout, int Coutp, int Khalf, int Khalfp) {
    const int KPAD = 2 * Khalfp;
    int i = blockIdx.x * blockDim.x + threadIdx.x;
    if (i >= Coutp * KPAD) return;
    int o    = i / KPAD;
    int khw  = i - o * KPAD;
    int chnk = khw >> 5;
    int r    = khw & 31;
    int half = r >> 4;
    int t    = r & 15;
    int sidx = chnk * 16 + t;
    f16 v = (f16)0;
    if (o < Cout && sidx < Khalf) {
        // K (logical, per out-row) = 2*Khalf = CIN*KS*KS
        v = (f16)w[(size_t)o * (2 * Khalf) + half * Khalf + sidx];
    }
    wpk[i] = v;
}

// ---------------------------------------------------------------------------
// Build conv1 input: [B][20][H][W] f16. ch 0-11 = frame f of x, ch 12-19 =
// recurrent hidden = channels 0..7 of previous frame's `raw` (zero at f==0).
// ---------------------------------------------------------------------------
__global__ void pack_input(const float* __restrict__ x, const f16* __restrict__ raw,
                           f16* __restrict__ inp, int f, int HW) {
    int i = blockIdx.x * blockDim.x + threadIdx.x;
    if (i >= 2 * 20 * HW) return;
    int b = i / (20 * HW);
    int r = i - b * 20 * HW;
    int c = r / HW;
    int p = r - c * HW;
    f16 v;
    if (c < 12)       v = (f16)x[((size_t)b * 48 + (size_t)f * 12 + c) * HW + p];
    else if (f == 0)  v = (f16)0;
    else              v = raw[((size_t)b * 24 + (c - 12)) * HW + p];
    inp[i] = v;
}

// ---------------------------------------------------------------------------
// WMMA implicit-GEMM conv, reflect padding.
//   M = out channels (16-wide tiles), N = 16 consecutive pixels in a row,
//   K = CIN*KSZ*KSZ (two channel streams, each padded to 16). Block = 256 thr
//   (8 waves) = one 64-pixel row strip x all COUTP channels. Input patch
//   staged in LDS; chunk loop fully unrolled so all B offsets are immediates.
// ---------------------------------------------------------------------------
template <int CIN, int COUT, int COUTP, int KSZ, int PAD, bool RELU>
__global__ __launch_bounds__(256)
void conv_wmma(const f16* __restrict__ in, const f16* __restrict__ wpk,
               const float* __restrict__ bias, f16* __restrict__ out,
               int Wd, int Hd) {
    constexpr int KHALF  = (CIN / 2) * KSZ * KSZ;   // taps per channel stream
    constexpr int KHALFP = (KHALF + 15) & ~15;
    constexpr int KPAD   = 2 * KHALFP;              // hw-K extent per weight row
    constexpr int CHUNKS = KHALFP / 16;             // 32-wide hw-K chunks
    constexpr int TW     = 64 + KSZ - 1;            // strip + halo
    constexpr int TWP    = (TW + 7) & ~7;           // padded pitch
    constexpr int MT     = COUTP / 16;

    __shared__ f16 smem[CIN * KSZ * TWP];

    const int tid  = threadIdx.x;
    const int lane = tid & 31;
    const int wv   = tid >> 5;
    const int x0   = blockIdx.x * 64;
    const int y    = blockIdx.y;
    const int b    = blockIdx.z;
    const size_t HWs = (size_t)Wd * Hd;
    const f16* inb = in + (size_t)b * CIN * HWs;

    // Warm weight rows in L2 (global_prefetch_b8)
    __builtin_prefetch(wpk + (size_t)lane * KPAD, 0, 3);

    // Cooperative LDS staging of the reflect-padded patch (CIN x KSZ x TW)
    for (int e = tid; e < CIN * KSZ * TW; e += 256) {
        int cin = e / (KSZ * TW);
        int r   = e - cin * (KSZ * TW);
        int ky  = r / TW;
        int tx  = r - ky * TW;
        int iy  = reflect_idx(y + ky - PAD, Hd);
        int ix  = reflect_idx(x0 - PAD + tx, Wd);
        smem[(cin * KSZ + ky) * TWP + tx] = inb[(size_t)cin * HWs + (size_t)iy * Wd + ix];
    }
    __syncthreads();

    const int halfsel = lane >> 4;   // K-half selector (contiguous k order used by this code)
    const int mn      = lane & 15;   // A row (out channel) == B column (pixel)
    // Per-lane stream base: lane half h reads channel stream h.
    const f16* sb = smem + halfsel * ((CIN / 2) * KSZ * TWP);

    for (int tile = wv; tile < MT * 4; tile += 8) {
        const int m0 = (tile >> 2) * 16;
        const int p0 = (tile & 3) * 16;
        v8f acc = {};
        const f16* wrow = wpk + (size_t)(m0 + mn) * KPAD + halfsel * 16;
        const f16* sp   = sb + p0 + mn;   // all B loads: sp[compile-time imm]

#pragma unroll
        for (int c = 0; c < CHUNKS; ++c) {
            const int kc = c * 32;
            // A fragment: lanes 0-15 hw-K=(kc+0..15),
            //             lanes 16-31 hw-K=(kc+16..31)
            v8h alo = *(const v8h*)(wrow + kc);
            v8h ahi = *(const v8h*)(wrow + kc + 8);
            v16h a  = __builtin_shufflevector(alo, ahi,
                        0, 1, 2, 3, 4, 5, 6, 7, 8, 9, 10, 11, 12, 13, 14, 15);

            // B fragment: element t = stream tap (c*16 + t); offsets fold to
            // ds_load_u16 immediates after unroll.
            v16h bm;
#pragma unroll
            for (int t = 0; t < 16; ++t) {
                const int sidx = c * 16 + t;
                if (sidx < KHALF) {
                    const int cin = sidx / (KSZ * KSZ);
                    const int rr  = sidx - cin * (KSZ * KSZ);
                    const int ky  = rr / KSZ;
                    const int kx  = rr - ky * KSZ;
                    bm[t] = sp[(cin * KSZ + ky) * TWP + kx];
                } else {
                    bm[t] = (f16)0;
                }
            }
            acc = __builtin_amdgcn_wmma_f32_16x16x32_f16(
                      false, a, false, bm, (short)0, acc, false, false);
        }

        // D layout: lane -> N=lane&15 (pixel), VGPR j -> M=j+8*halfsel
        const int px = x0 + p0 + mn;
#pragma unroll
        for (int j = 0; j < 8; ++j) {
            int row = m0 + j + halfsel * 8;
            if (row < COUT) {
                float v = acc[j] + bias[row];
                if (RELU) v = v > 0.f ? v : 0.f;
                out[((size_t)b * COUT + row) * HWs + (size_t)y * Wd + px] = (f16)v;
            }
        }
    }
}

// ---------------------------------------------------------------------------
// Tail: softmax(raw[8:24]/5) -> per-pixel 9x9 blended filter over color,
// multiplied by albedo. Reassociated: filtered_c = sum_taps color_c * wsum,
// wsum = sum_k prob_k * dw[k][tap]. Zero padding (matches lax explicit pad).
// ---------------------------------------------------------------------------
__global__ __launch_bounds__(256)
void postproc(const float* __restrict__ x, const f16* __restrict__ raw,
              const float* __restrict__ dwk, float* __restrict__ outp,
              int f, int Wd, int Hd) {
    __shared__ float sdw[16 * 81];
    const int tid = threadIdx.x;
    for (int i = tid; i < 16 * 81; i += 256) sdw[i] = dwk[i];
    __syncthreads();

    const size_t HW = (size_t)Wd * Hd;
    const size_t p  = (size_t)blockIdx.x * 256 + tid;
    const int b     = blockIdx.y;
    if (p >= HW) return;
    const int yy = (int)(p / Wd);
    const int xx = (int)(p - (size_t)yy * Wd);

    // softmax over 16 kernel logits / 5
    float lg[16];
    float mx = -1e30f;
#pragma unroll
    for (int k = 0; k < 16; ++k) {
        lg[k] = 0.2f * (float)raw[((size_t)b * 24 + 8 + k) * HW + p];
        mx = fmaxf(mx, lg[k]);
    }
    float s = 0.f;
#pragma unroll
    for (int k = 0; k < 16; ++k) { lg[k] = __expf(lg[k] - mx); s += lg[k]; }
    const float inv = 1.f / s;
#pragma unroll
    for (int k = 0; k < 16; ++k) lg[k] *= inv;

    const float* col = x + ((size_t)b * 48 + (size_t)f * 12) * HW;  // color 0..2
    float a0 = 0.f, a1 = 0.f, a2 = 0.f;
    for (int dy = 0; dy < 9; ++dy) {
        const int iy = yy + dy - 4;
        if (iy < 0 || iy >= Hd) continue;
        for (int dx = 0; dx < 9; ++dx) {
            const int ix = xx + dx - 4;
            if (ix < 0 || ix >= Wd) continue;
            const int t = dy * 9 + dx;
            float wsum = 0.f;
#pragma unroll
            for (int k = 0; k < 16; ++k) wsum += lg[k] * sdw[k * 81 + t];
            const size_t q = (size_t)iy * Wd + ix;
            a0 += wsum * col[q];
            a1 += wsum * col[HW + q];
            a2 += wsum * col[2 * HW + q];
        }
    }
    const float* alb = col + 3 * HW;                      // albedo 3..5
    float* ob = outp + ((size_t)b * 12 + (size_t)f * 3) * HW;
    ob[p]          = alb[p]          * a0;
    ob[HW + p]     = alb[HW + p]     * a1;
    ob[2 * HW + p] = alb[2 * HW + p] * a2;
}

// ---------------------------------------------------------------------------
extern "C" void kernel_launch(void* const* d_in, const int* in_sizes, int n_in,
                              void* d_out, int out_size, void* d_ws, size_t ws_size,
                              hipStream_t stream) {
    (void)in_sizes; (void)n_in; (void)out_size; (void)ws_size;
    const float* x   = (const float*)d_in[0];
    const float* w1  = (const float*)d_in[1];
    const float* b1  = (const float*)d_in[2];
    const float* w2  = (const float*)d_in[3];
    const float* b2  = (const float*)d_in[4];
    const float* w3  = (const float*)d_in[5];
    const float* b3  = (const float*)d_in[6];
    const float* w4  = (const float*)d_in[7];
    const float* b4  = (const float*)d_in[8];
    const float* dwk = (const float*)d_in[9];
    float* outp = (float*)d_out;

    const int Wd = 512, Hd = 512;
    const size_t HW = (size_t)Wd * Hd;

    // Workspace (f16 elements), ping-pong aliased:
    //   regionA = inp(20ch) / h2(32ch)   : 2*32*HW
    //   regionB = h1(16ch)  / h3(64ch)   : 2*64*HW
    //   raw     = conv4 out (24ch; ch0-7 doubles as recurrent hidden)
    //   packed f16 weights after that (16B-aligned rows)
    f16* regionA = (f16*)d_ws;
    f16* regionB = regionA + (size_t)2 * 32 * HW;
    f16* raw     = regionB + (size_t)2 * 64 * HW;
    f16* wpk1    = raw     + (size_t)2 * 24 * HW;
    f16* wpk2    = wpk1 + (size_t)16 * 992;    // Khalf=490 -> 496, KPAD=992
    f16* wpk3    = wpk2 + (size_t)32 * 416;    // Khalf=200 -> 208, KPAD=416
    f16* wpk4    = wpk3 + (size_t)64 * 800;    // Khalf=400 -> 400, KPAD=800
    f16* inp = regionA;                        // Khalf=800 -> 800, KPAD=1600
    f16* h1  = regionB;
    f16* h2  = regionA;
    f16* h3  = regionB;

    // Weight repacks with the hw-K permutation (deterministic every call)
    pack_weights<<<(16 * 992  + 255) / 256, 256, 0, stream>>>(w1, wpk1, 16, 16, 490, 496);
    pack_weights<<<(32 * 416  + 255) / 256, 256, 0, stream>>>(w2, wpk2, 32, 32, 200, 208);
    pack_weights<<<(64 * 800  + 255) / 256, 256, 0, stream>>>(w3, wpk3, 64, 64, 400, 400);
    pack_weights<<<(32 * 1600 + 255) / 256, 256, 0, stream>>>(w4, wpk4, 24, 32, 800, 800);

    const dim3 cgrid(Wd / 64, Hd, 2);
    const dim3 pgrid((unsigned)((HW + 255) / 256), 2);

    for (int f = 0; f < 4; ++f) {
        pack_input<<<(unsigned)((2 * 20 * HW + 255) / 256), 256, 0, stream>>>(
            x, raw, inp, f, (int)HW);
        conv_wmma<20, 16, 16, 7, 3, true ><<<cgrid, 256, 0, stream>>>(inp, wpk1, b1, h1, Wd, Hd);
        conv_wmma<16, 32, 32, 5, 2, true ><<<cgrid, 256, 0, stream>>>(h1,  wpk2, b2, h2, Wd, Hd);
        conv_wmma<32, 64, 64, 5, 2, true ><<<cgrid, 256, 0, stream>>>(h2,  wpk3, b3, h3, Wd, Hd);
        conv_wmma<64, 24, 32, 5, 2, false><<<cgrid, 256, 0, stream>>>(h3,  wpk4, b4, raw, Wd, Hd);
        postproc<<<pgrid, 256, 0, stream>>>(x, raw, dwk, outp, f, Wd, Hd);
    }
}
